// CausalSelfAttention_80848464380125
// MI455X (gfx1250) — matrix operand order repaired and checked by reading, not compiled, not run on hardware
//
#include <hip/hip_runtime.h>

// ---------------------------------------------------------------------------
// CDNA5 (gfx1250) causal self-attention, bf16 WMMA pipeline, f32 accumulate.
// B=2, L=2048, H=16, D=64, HID=1024.
// Data movement: GLOBAL_LOAD_ASYNC_TO_LDS_B128 for row-major tiles (ASYNCcnt),
// VGPR-path transposed staging for B/V tiles, s_wait_dscnt for wave-local
// LDS relayout in the softmax->PV step.
// ---------------------------------------------------------------------------

typedef __attribute__((ext_vector_type(16))) __bf16 v16bf;
typedef __attribute__((ext_vector_type(8)))  float  v8f;

#define CB   2
#define CL   2048
#define CH   16
#define CD   64
#define CHID 1024
#define CN3  3072
#define CEPS 1e-6f

union FragAB { unsigned u[8]; v16bf v; };

__device__ __forceinline__ unsigned short f2bf(float f) {
    return __builtin_bit_cast(unsigned short, (__bf16)f);
}

// uint index (bf16 pairs) within a 16-uint span for the WMMA A-operand layout
// (16x32, 16-bit): lanes 0-15 K=0..7 / 16..23, lanes 16-31 K=8..15 / 24..31.
__device__ __forceinline__ int apair(int g, int p) {
    return (p < 4) ? (g * 4 + p) : (8 + g * 4 + (p - 4));
}

__device__ __forceinline__ float red_max16(float v) {
    #pragma unroll
    for (int o = 1; o < 16; o <<= 1) v = fmaxf(v, __shfl_xor(v, o, 32));
    return v;
}
__device__ __forceinline__ float red_sum16(float v) {
    #pragma unroll
    for (int o = 1; o < 16; o <<= 1) v += __shfl_xor(v, o, 32);
    return v;
}

// Async DMA: 16B global -> LDS. IOFFSET is added to BOTH the LDS and global
// address (ISA 08_async_tensor §4.4), so one (lds, gaddr) pair serves 4 chunks.
#define GLD_ASYNC_B128(ldsoff, gaddr, IMM)                                    \
    asm volatile("global_load_async_to_lds_b128 %0, %1, off offset:" IMM      \
                 :: "v"(ldsoff), "v"(gaddr) : "memory")
#define WAIT_ASYNC0() asm volatile("s_wait_asynccnt 0x0" ::: "memory")
#define WAIT_DS0()    asm volatile("s_wait_dscnt 0x0" ::: "memory")

__device__ __forceinline__ unsigned lds_off(const void* p) {
    // Generic LDS address: addr[31:0] is the wave-relative LDS byte offset.
    return (unsigned)(unsigned long long)p;
}

// ---------------------------------------------------------------------------
// Kernel: f32 -> bf16 elementwise convert
// ---------------------------------------------------------------------------
__global__ __launch_bounds__(256) void k_cvt_bf16(const float* __restrict__ in,
                                                  unsigned short* __restrict__ out,
                                                  int n) {
    int i = blockIdx.x * 256 + threadIdx.x;
    if (i < n) out[i] = f2bf(in[i]);
}

// ---------------------------------------------------------------------------
// Kernel: C(f32, MxN) = A(bf16, MxK) * B(bf16, KxN), row-major. N,K static.
// Block tile 128x128, BK=64. 8 waves in 4x2: each wave 32 rows x 64 cols
// (8 C tiles, 16 WMMAs per staging round).
// A tile staged with async-to-LDS B128 DMA; B tile staged transposed.
// ---------------------------------------------------------------------------
template <int N, int K>
__global__ __launch_bounds__(256) void k_gemm_bf16(const unsigned short* __restrict__ A,
                                                   const unsigned short* __restrict__ Bm,
                                                   float* __restrict__ C) {
    __shared__ unsigned Asu[128 * 36];   // 128 rows x 64 bf16, 144B row stride (16B aligned)
    __shared__ unsigned Bsu[128 * 33];   // transposed: 128 cols x 64 k, 33-uint col stride

    const int tid  = threadIdx.x;
    const int lane = tid & 31;
    const int w    = tid >> 5;
    const int g    = lane >> 4;
    const int n    = lane & 15;
    const int m    = lane & 15;
    const int m0   = (w & 3) * 32;      // wave row origin (2 stripes of 16)
    const int n0   = (w >> 2) * 64;     // wave col origin (4 tiles of 16)
    const int blkM = blockIdx.y * 128;
    const int blkN = blockIdx.x * 128;

    // --- staging addresses (all loop-invariant; one bump per k-iter) ---
    // A: thread covers (row = tid>>1, 64B half = tid&1), 4 async B128 chunks.
    const int aRow = tid >> 1, aHalf = (tid & 1) * 64;
    unsigned long long aG =
        (unsigned long long)(const char*)(A + (size_t)(blkM + aRow) * K) + aHalf;
    const unsigned aL = lds_off(Asu) + aRow * 144 + aHalf;
    // B: thread covers (krBase = tid>>6, uint col cp = tid&63), 16 rows apart by 4.
    const int krB = tid >> 6, cp = tid & 63;
    const unsigned* pB = (const unsigned*)Bm + (size_t)krB * (N / 2) + (blkN >> 1) + cp;
    unsigned short* bs16 = (unsigned short*)Bsu;
    const int bs0 = (2 * cp) * 66 + krB;

    v8f acc[2][4] = {};

    for (int kit = 0; kit < K / 64; ++kit) {
        __syncthreads();
        // A tile: 4 x async B128 (IOFFSET shifts both LDS and global address)
        GLD_ASYNC_B128(aL, aG, "0");
        GLD_ASYNC_B128(aL, aG, "16");
        GLD_ASYNC_B128(aL, aG, "32");
        GLD_ASYNC_B128(aL, aG, "48");
        aG += 128;
        // B tile: 16 uint loads (immediate offsets) -> 32 ds_store_b16 transposed
        #pragma unroll
        for (int i = 0; i < 16; i++) {
            unsigned u = pB[(size_t)i * 2 * N];
            bs16[bs0 + 4 * i]      = (unsigned short)u;
            bs16[bs0 + 66 + 4 * i] = (unsigned short)(u >> 16);
        }
        pB += (size_t)32 * N;
        if (kit + 1 < K / 64) __builtin_prefetch(pB, 0, 1);
        WAIT_ASYNC0();
        __syncthreads();

        #pragma unroll
        for (int ks = 0; ks < 2; ks++) {           // two K=32 sub-steps
            FragAB fa[2];
            #pragma unroll
            for (int s = 0; s < 2; s++)
                #pragma unroll
                for (int p = 0; p < 8; p++)
                    fa[s].u[p] = Asu[(m0 + s * 16 + m) * 36 + ks * 16 + g * 8 + p];
            #pragma unroll
            for (int t = 0; t < 4; t++) {
                FragAB fb;
                #pragma unroll
                for (int p = 0; p < 8; p++)
                    fb.u[p] = Bsu[(n0 + t * 16 + n) * 33 + ks * 16 + g * 8 + p];
                #pragma unroll
                for (int s = 0; s < 2; s++)
                    acc[s][t] = __builtin_amdgcn_wmma_f32_16x16x32_bf16(
                        false, fa[s].v, false, fb.v, (short)0, acc[s][t], false, false);
            }
        }
    }

    #pragma unroll
    for (int s = 0; s < 2; s++)
        #pragma unroll
        for (int t = 0; t < 4; t++)
            #pragma unroll
            for (int r = 0; r < 8; r++) {
                int row = blkM + m0 + s * 16 + r + 8 * g;   // C: VGPR r -> M=r+8*(lane/16)
                int col = blkN + n0 + t * 16 + n;
                C[(size_t)row * N + col] = acc[s][t][r];
            }
}

// ---------------------------------------------------------------------------
// Kernel: RMSNorm(q,k) + RoPE(q,k) + repack q/k/v -> [B,H,L,D] bf16.
// ---------------------------------------------------------------------------
__global__ __launch_bounds__(64) void k_normrope(const float* __restrict__ qkv,
                                                 const float* __restrict__ qw,
                                                 const float* __restrict__ kw,
                                                 const float* __restrict__ cosb,
                                                 const float* __restrict__ sinb,
                                                 unsigned short* __restrict__ qb,
                                                 unsigned short* __restrict__ kb,
                                                 unsigned short* __restrict__ vb) {
    const int id = blockIdx.x;
    const int h  = id % CH;
    const int l  = (id / CH) % CL;
    const int b  = id / (CH * CL);
    const int d  = threadIdx.x;

    __shared__ float red[64], tq[64], tk[64];

    const size_t base = ((size_t)(b * CL + l) * 3) * CHID + h * CD + d;
    const float q = qkv[base];
    const float k = qkv[base + CHID];
    const float v = qkv[base + 2 * CHID];

    red[d] = q * q; __syncthreads();
    for (int s = 32; s > 0; s >>= 1) { if (d < s) red[d] += red[d + s]; __syncthreads(); }
    const float rq = rsqrtf(red[0] * (1.0f / CD) + CEPS);
    __syncthreads();
    red[d] = k * k; __syncthreads();
    for (int s = 32; s > 0; s >>= 1) { if (d < s) red[d] += red[d + s]; __syncthreads(); }
    const float rk = rsqrtf(red[0] * (1.0f / CD) + CEPS);

    tq[d] = q * rq * qw[d];
    tk[d] = k * rk * kw[d];
    __syncthreads();

    float oq, ok;
    if (d < 32) {
        const float c = cosb[l * 32 + d], s = sinb[l * 32 + d];
        oq = tq[d] * c - tq[d + 32] * s;
        ok = tk[d] * c - tk[d + 32] * s;
    } else {
        const float c = cosb[l * 32 + d - 32], s = sinb[l * 32 + d - 32];
        oq = tq[d - 32] * s + tq[d] * c;
        ok = tk[d - 32] * s + tk[d] * c;
    }

    const size_t ob = (((size_t)b * CH + h) * CL + l) * CD + d;
    qb[ob] = f2bf(oq);
    kb[ob] = f2bf(ok);
    vb[ob] = f2bf(v);
}

// ---------------------------------------------------------------------------
// Kernel: causal flash attention. Block = (b, h, 128 q rows), 8 waves.
// K tile staged via async-to-LDS DMA, V tile transposed via VGPR path.
// S = Q*K^T (4 WMMAs / 32 keys), online softmax in C-layout f32 regs,
// P relayout through wave-private LDS (s_wait_dscnt), O += P*V (4 WMMAs).
// ---------------------------------------------------------------------------
__global__ __launch_bounds__(256) void k_attn(const unsigned short* __restrict__ qb,
                                              const unsigned short* __restrict__ kb,
                                              const unsigned short* __restrict__ vb,
                                              unsigned short* __restrict__ attnb) {
    __shared__ unsigned shKt[32 * 36];     // 32 keys x 64 d bf16, 144B stride (16B aligned)
    __shared__ unsigned shVt[64 * 17];     // transposed: 64 d x 32 keys, 17-uint stride
    __shared__ unsigned shP[8][256];       // per-wave P tile: 16 rows x 32 cols bf16

    const int id   = blockIdx.x;
    const int qblk = id & 15;
    const int h    = (id >> 4) & 15;
    const int b    = id >> 8;

    const int tid  = threadIdx.x;
    const int lane = tid & 31;
    const int w    = tid >> 5;
    const int g    = lane >> 4;
    const int n    = lane & 15;
    const int m    = lane & 15;

    const int qb0 = qblk * 128;
    const int qw0 = qb0 + w * 16;
    const size_t head = ((size_t)b * CH + h) * CL;

    // Q fragments (d 0..31, 32..63) for this wave's 16 rows
    FragAB qf0, qf1;
    {
        const unsigned* qrow = (const unsigned*)(qb + (head + qw0 + m) * CD);
        #pragma unroll
        for (int p = 0; p < 8; p++) {
            const int ip = g * 8 + p;
            qf0.u[p] = qrow[ip];
            qf1.u[p] = qrow[16 + ip];
        }
    }

    // staging pointers (one bump per 32-key step)
    const int kRow = tid >> 3, kChunk = (tid & 7) * 16;        // K: 1 async B128/thread
    unsigned long long kG =
        (unsigned long long)(const char*)(kb + (head + kRow) * CD) + kChunk;
    const unsigned kL = lds_off(shKt) + kRow * 144 + kChunk;
    const int vKey = tid >> 5, vcp = tid & 31;                 // V: 4 uints/thread
    const unsigned* vG = (const unsigned*)(vb + (head + vKey) * CD) + vcp;
    unsigned short* vt16 = (unsigned short*)shVt;
    const int vs0 = (2 * vcp) * 34 + vKey;

    v8f acc[4] = {};
    float mrun[8], lrun[8];
    #pragma unroll
    for (int r = 0; r < 8; r++) { mrun[r] = -1e30f; lrun[r] = 0.0f; }
    const float scale = 0.125f;  // 1/sqrt(64)

    for (int k0 = 0; k0 < qb0 + 128; k0 += 32) {
        __syncthreads();
        GLD_ASYNC_B128(kL, kG, "0");                 // K tile DMA
        kG += 32 * CD * 2;
        #pragma unroll
        for (int i = 0; i < 4; i++) {                // V tile transposed
            unsigned uv = vG[i * 256];
            vt16[vs0 + 8 * i]      = (unsigned short)uv;
            vt16[vs0 + 34 + 8 * i] = (unsigned short)(uv >> 16);
        }
        vG += 1024;
        WAIT_ASYNC0();
        __syncthreads();

        // S tiles: keys [k0,k0+16) and [k0+16,k0+32)
        v8f S0 = {}, S1 = {};
        FragAB kf;
        #pragma unroll
        for (int p = 0; p < 8; p++) kf.u[p] = shKt[n * 36 + g * 8 + p];
        S0 = __builtin_amdgcn_wmma_f32_16x16x32_bf16(false, qf0.v, false, kf.v, (short)0, S0, false, false);
        #pragma unroll
        for (int p = 0; p < 8; p++) kf.u[p] = shKt[n * 36 + 16 + g * 8 + p];
        S0 = __builtin_amdgcn_wmma_f32_16x16x32_bf16(false, qf1.v, false, kf.v, (short)0, S0, false, false);
        #pragma unroll
        for (int p = 0; p < 8; p++) kf.u[p] = shKt[(16 + n) * 36 + g * 8 + p];
        S1 = __builtin_amdgcn_wmma_f32_16x16x32_bf16(false, qf0.v, false, kf.v, (short)0, S1, false, false);
        #pragma unroll
        for (int p = 0; p < 8; p++) kf.u[p] = shKt[(16 + n) * 36 + 16 + g * 8 + p];
        S1 = __builtin_amdgcn_wmma_f32_16x16x32_bf16(false, qf1.v, false, kf.v, (short)0, S1, false, false);

        // online softmax (rows live per-VGPR in the C layout)
        unsigned short* pw = (unsigned short*)shP[w];
        #pragma unroll
        for (int r = 0; r < 8; r++) {
            const int qi = qw0 + r + 8 * g;
            float s0 = S0[r] * scale; if (k0 + n      > qi) s0 = -1e30f;
            float s1 = S1[r] * scale; if (k0 + 16 + n > qi) s1 = -1e30f;
            const float mt = red_max16(fmaxf(s0, s1));
            const float mn = fmaxf(mrun[r], mt);
            const float al = __expf(mrun[r] - mn);
            const float p0 = __expf(s0 - mn);
            const float p1 = __expf(s1 - mn);
            lrun[r] = lrun[r] * al + red_sum16(p0 + p1);
            mrun[r] = mn;
            #pragma unroll
            for (int t = 0; t < 4; t++) acc[t][r] *= al;
            pw[(r + 8 * g) * 32 + n]      = f2bf(p0);
            pw[(r + 8 * g) * 32 + 16 + n] = f2bf(p1);
        }
        WAIT_DS0();   // wave-private C-layout -> A-layout relayout via LDS

        FragAB pf;
        #pragma unroll
        for (int p = 0; p < 8; p++) pf.u[p] = shP[w][m * 16 + g * 8 + p];
        #pragma unroll
        for (int t = 0; t < 4; t++) {
            FragAB vf;
            #pragma unroll
            for (int p = 0; p < 8; p++) vf.u[p] = shVt[(t * 16 + n) * 17 + g * 8 + p];
            acc[t] = __builtin_amdgcn_wmma_f32_16x16x32_bf16(
                false, pf.v, false, vf.v, (short)0, acc[t], false, false);
        }
    }

    // epilogue: O /= l -> bf16 [B, L, HID]
    #pragma unroll
    for (int r = 0; r < 8; r++) {
        const int qi = qw0 + r + 8 * g;
        const float inv = 1.0f / lrun[r];
        #pragma unroll
        for (int t = 0; t < 4; t++)
            attnb[(size_t)(b * CL + qi) * CHID + h * CD + t * 16 + n] =
                f2bf(acc[t][r] * inv);
    }
}

// ---------------------------------------------------------------------------
// Host launcher
// ---------------------------------------------------------------------------
extern "C" void kernel_launch(void* const* d_in, const int* in_sizes, int n_in,
                              void* d_out, int out_size, void* d_ws, size_t ws_size,
                              hipStream_t stream) {
    const float* x    = (const float*)d_in[0];
    const float* Wqkv = (const float*)d_in[1];
    const float* Wo   = (const float*)d_in[2];
    const float* qw   = (const float*)d_in[3];
    const float* kw   = (const float*)d_in[4];
    const float* cosb = (const float*)d_in[5];
    const float* sinb = (const float*)d_in[6];
    float* out = (float*)d_out;

    const int M    = CB * CL;            // 4096
    const int nX   = M * CHID;           // 4,194,304
    const int nWq  = CHID * CN3;         // 3,145,728
    const int nWo  = CHID * CHID;        // 1,048,576
    const int nQKV = M * CN3;            // 12,582,912 f32
    const int nHd  = CB * CH * CL * CD;  // 4,194,304

    char* ws = (char*)d_ws;
    size_t off = 0;
    unsigned short* xb    = (unsigned short*)(ws + off); off += (size_t)nX  * 2;
    unsigned short* wqkvb = (unsigned short*)(ws + off); off += (size_t)nWq * 2;
    unsigned short* wob   = (unsigned short*)(ws + off); off += (size_t)nWo * 2;
    float*          qkvf  = (float*)         (ws + off); off += (size_t)nQKV * 4;
    unsigned short* qbuf  = (unsigned short*)(ws + off); off += (size_t)nHd * 2;
    unsigned short* kbuf  = (unsigned short*)(ws + off); off += (size_t)nHd * 2;
    unsigned short* vbuf  = (unsigned short*)(ws + off); off += (size_t)nHd * 2;
    unsigned short* attnb = (unsigned short*)(ws + off); off += (size_t)nX  * 2;
    (void)ws_size; (void)in_sizes; (void)n_in; (void)out_size;

    k_cvt_bf16<<<(nX  + 255) / 256, 256, 0, stream>>>(x,    xb,    nX);
    k_cvt_bf16<<<(nWq + 255) / 256, 256, 0, stream>>>(Wqkv, wqkvb, nWq);
    k_cvt_bf16<<<(nWo + 255) / 256, 256, 0, stream>>>(Wo,   wob,   nWo);

    k_gemm_bf16<CN3, CHID><<<dim3(CN3 / 128, M / 128), 256, 0, stream>>>(xb, wqkvb, qkvf);

    k_normrope<<<CB * CL * CH, 64, 0, stream>>>(qkvf, qw, kw, cosb, sinb, qbuf, kbuf, vbuf);

    k_attn<<<CB * CH * (CL / 128), 256, 0, stream>>>(qbuf, kbuf, vbuf, attnb);

    k_gemm_bf16<CHID, CHID><<<dim3(CHID / 128, M / 128), 256, 0, stream>>>(attnb, wob, out);
}
